// MultiHeadAttention_24257975288547
// MI455X (gfx1250) — matrix operand order repaired and checked by reading, not compiled, not run on hardware
//
#include <hip/hip_runtime.h>
#include <hip/hip_bf16.h>

// ---------------------------------------------------------------------------
// MHA for MI455X (gfx1250): bf16 WMMA (v_wmma_f32_16x16x32_bf16) everywhere,
// fp32 accumulation, fp32 softmax. wave32 / WGP model.
// GEMM uses async Global->LDS staging (ASYNCcnt) with double buffering.
// ---------------------------------------------------------------------------

#define HH   16
#define SS   1024
#define DK   64
#define DM   1024
#define BB   4

typedef __attribute__((ext_vector_type(16))) __bf16 v16bf;
typedef __attribute__((ext_vector_type(8)))  float  v8f;

static __device__ __forceinline__ v8f vzero8() {
  v8f z = {0.f, 0.f, 0.f, 0.f, 0.f, 0.f, 0.f, 0.f};
  return z;
}

static __device__ __forceinline__ v8f wmma_bf16(v16bf a, v16bf b, v8f c) {
  // args: (neg_a, A, neg_b, B, c_mod, C, reuse_a, reuse_b)
  return __builtin_amdgcn_wmma_f32_16x16x32_bf16(false, a, false, b, (short)0, c,
                                                 false, false);
}

// A-fragment (16x32 bf16, MxK): lane L<16 holds row L, K = k0+{0..15};
// lane L>=16 holds row L-16, K = k0+{16..31}.  Two 16B loads.
static __device__ __forceinline__ v16bf ldA(const __hip_bfloat16* A, int ld,
                                            int row0, int k0, int lane) {
  int r = row0 + (lane & 15);
  const char* p = (const char*)(A + (size_t)r * ld + k0 + ((lane >> 4) << 4));
  union { v16bf v; uint4 u[2]; } f;
  f.u[0] = *(const uint4*)p;
  f.u[1] = *(const uint4*)(p + 16);   // +8 elements
  return f.v;
}

// B-fragment (32x16 bf16, KxN) for NT-GEMM: operand stored row-major [N,K],
// lane L holds column N=col0+(L%16), K = k0 + (L/16)*16 + {0..15} contiguous.
static __device__ __forceinline__ v16bf ldB(const __hip_bfloat16* Bm, int ld,
                                            int col0, int k0, int lane) {
  int c = col0 + (lane & 15);
  const char* p = (const char*)(Bm + (size_t)c * ld + k0 + ((lane >> 4) << 4));
  union { v16bf v; uint4 u[2]; } f;
  f.u[0] = *(const uint4*)p;
  f.u[1] = *(const uint4*)(p + 16);
  return f.v;
}

// LDS A-fragment, row-major slab with leading dim ld (elements)
static __device__ __forceinline__ v16bf ldA_lds(const __hip_bfloat16* sp, int ld,
                                                int row0, int k0, int lane) {
  int r = row0 + (lane & 15);
  const __hip_bfloat16* p = sp + (size_t)r * ld + k0 + ((lane >> 4) << 4);
  union { v16bf v; uint4 u[2]; } f;
  f.u[0] = *(const uint4*)p;
  f.u[1] = *(const uint4*)(p + 8);   // +8 bf16 elements
  return f.v;
}

// LDS B-fragment, operand stored row-major [N,ld] in LDS
static __device__ __forceinline__ v16bf ldB_lds(const __hip_bfloat16* sp, int ld,
                                                int col0, int k0, int lane) {
  int c = col0 + (lane & 15);
  const __hip_bfloat16* p = sp + (size_t)c * ld + k0 + ((lane >> 4) << 4);
  union { v16bf v; uint4 u[2]; } f;
  f.u[0] = *(const uint4*)p;
  f.u[1] = *(const uint4*)(p + 8);
  return f.v;
}

// Async global -> LDS, 16 bytes per lane; tracked by ASYNCcnt.
static __device__ __forceinline__ void async_g2l_b128(unsigned lds_byte_addr,
                                                      const void* gptr) {
  unsigned long long ga = (unsigned long long)gptr;
  asm volatile("global_load_async_to_lds_b128 %0, %1, off"
               :: "v"(lds_byte_addr), "v"(ga) : "memory");
}
static __device__ __forceinline__ void wait_async0() {
  asm volatile("s_wait_asynccnt 0x0" ::: "memory");
}

// ---------------------------------------------------------------------------
// fp32 -> bf16 convert (memory bound; whole pass ~ few us at 23.3 TB/s)
// ---------------------------------------------------------------------------
__global__ void f2bf_kernel(const float* __restrict__ src,
                            __hip_bfloat16* __restrict__ dst, int n) {
  int i = blockIdx.x * blockDim.x + threadIdx.x;
  int stride = gridDim.x * blockDim.x;
  for (; i < n; i += stride) dst[i] = __float2bfloat16(src[i]);
}

// ---------------------------------------------------------------------------
// NT-GEMM: C[m,n] = sum_k A[m,k] * W[n,k] + bias[n]
//   A: [M,K] bf16 row-major, W: [N,K] bf16 row-major (nn.Linear weight)
//   mode 0: out bf16, layout [B,H,S,Dk]   (q / k buffers)
//   mode 1: out bf16, layout [B,H,Dk,S]   (v transposed for PV NT-GEMM)
//   mode 2: out fp32, row-major [M,N]     (final projection)
// Block: 256 thr = 8 waves; block tile 128x128; wave tile 64x32 (4x2 WMMA).
// K-slabs (128x32 of A and of W) are double-buffered in LDS via async copies.
// ---------------------------------------------------------------------------
__global__ __launch_bounds__(256)
void gemm_nt_bf16(const __hip_bfloat16* __restrict__ A,
                  const __hip_bfloat16* __restrict__ W,
                  const float* __restrict__ bias,
                  void* __restrict__ out,
                  int M, int N, int K, int mode) {
  __shared__ __align__(16) __hip_bfloat16 sA[2][128 * 32];   // 2 x 8 KB
  __shared__ __align__(16) __hip_bfloat16 sB[2][128 * 32];   // 2 x 8 KB

  const int lane = threadIdx.x & 31;
  const int wid  = threadIdx.x >> 5;
  const int wy = wid >> 2, wx = wid & 3;
  const int rowblk = blockIdx.y * 128;
  const int colblk = blockIdx.x * 128;
  const int row0 = wy * 64;    // wave tile origin inside the LDS slab
  const int col0 = wx * 32;

  // Cooperative loader: each thread moves 2 x 16B chunks per slab per k-step.
  // Slab = 128 rows x 32 k (bf16, row-major). chunk c: row=c/4, kc=(c%4)*8.
  const int t = threadIdx.x;
  const int ra0 = t >> 2,        ka0 = (t & 3) * 8;
  const int ra1 = ra0 + 64,      ka1 = ka0;          // chunk t+256

  auto issue = [&](int buf, int k0) {
    async_g2l_b128((unsigned)(unsigned long long)&sA[buf][ra0 * 32 + ka0],
                   A + (size_t)(rowblk + ra0) * K + k0 + ka0);
    async_g2l_b128((unsigned)(unsigned long long)&sA[buf][ra1 * 32 + ka1],
                   A + (size_t)(rowblk + ra1) * K + k0 + ka1);
    async_g2l_b128((unsigned)(unsigned long long)&sB[buf][ra0 * 32 + ka0],
                   W + (size_t)(colblk + ra0) * K + k0 + ka0);
    async_g2l_b128((unsigned)(unsigned long long)&sB[buf][ra1 * 32 + ka1],
                   W + (size_t)(colblk + ra1) * K + k0 + ka1);
  };

  v8f acc[4][2];
  for (int i = 0; i < 4; ++i)
    for (int j = 0; j < 2; ++j) acc[i][j] = vzero8();

  const int nIt = K >> 5;   // K / 32
  int buf = 0;
  issue(0, 0);
  for (int it = 0; it < nIt; ++it) {
    wait_async0();        // this wave's slab data is in LDS
    __syncthreads();      // everyone's slab data is in LDS
    if (it + 1 < nIt) issue(buf ^ 1, (it + 1) * 32);

    v16bf bf0 = ldB_lds(&sB[buf][0], 32, col0,      0, lane);
    v16bf bf1 = ldB_lds(&sB[buf][0], 32, col0 + 16, 0, lane);
#pragma unroll
    for (int i = 0; i < 4; ++i) {
      v16bf af = ldA_lds(&sA[buf][0], 32, row0 + i * 16, 0, lane);
      acc[i][0] = wmma_bf16(af, bf0, acc[i][0]);
      acc[i][1] = wmma_bf16(af, bf1, acc[i][1]);
    }
    buf ^= 1;
  }

  // store: lane holds col = base + lane%16, rows base + (lane/16)*8 + e
#pragma unroll
  for (int i = 0; i < 4; ++i) {
#pragma unroll
    for (int j = 0; j < 2; ++j) {
      int c = colblk + col0 + j * 16 + (lane & 15);
      float bv = bias[c];
#pragma unroll
      for (int e = 0; e < 8; ++e) {
        int r = rowblk + row0 + i * 16 + ((lane >> 4) << 3) + e;
        float v = acc[i][j][e] + bv;
        if (mode == 2) {
          ((float*)out)[(size_t)r * N + c] = v;
        } else {
          int b = r >> 10, s = r & 1023;   // S = 1024
          int h = c >> 6,  d = c & 63;     // Dk = 64
          __hip_bfloat16* ob = (__hip_bfloat16*)out;
          if (mode == 0)
            ob[(((size_t)b * HH + h) * SS + s) * DK + d] = __float2bfloat16(v);
          else
            ob[(((size_t)b * HH + h) * DK + d) * SS + s] = __float2bfloat16(v);
        }
      }
    }
  }
}

// ---------------------------------------------------------------------------
// Attention: one block per (b, h, 32-row q tile). 8 waves.
//  stage 1: scores = 0.125 * q k^T, masked, -> bf16 in LDS (32x1024 = 64 KB)
//  stage 2: softmax in-place (8 lanes/row, __shfl_xor wave32 reductions)
//  stage 3: ctx = P @ v  (vT stored [B,H,Dk,S] => NT-GEMM, A-frags from LDS)
// ---------------------------------------------------------------------------
__global__ __launch_bounds__(256)
void attn_kernel(const __hip_bfloat16* __restrict__ qb,
                 const __hip_bfloat16* __restrict__ kb,
                 const __hip_bfloat16* __restrict__ vtb,
                 const int* __restrict__ mask,
                 __hip_bfloat16* __restrict__ ctx) {
  __shared__ __align__(16) __hip_bfloat16 sp[32 * 1024];   // 64 KB

  const int lane = threadIdx.x & 31;
  const int wid  = threadIdx.x >> 5;
  const int qt = blockIdx.x, h = blockIdx.y, b = blockIdx.z;
  const int bh = b * HH + h;
  const int q0 = qt * 32;

  const __hip_bfloat16* qh = qb  + (size_t)bh * SS * DK;
  const __hip_bfloat16* kh = kb  + (size_t)bh * SS * DK;
  const __hip_bfloat16* vh = vtb + (size_t)bh * DK * SS;
  const int* mb = mask + (size_t)b * SS * SS;

  // ---- stage 1: scores ----------------------------------------------------
  v16bf aq[2][2];
#pragma unroll
  for (int m = 0; m < 2; ++m)
#pragma unroll
    for (int t = 0; t < 2; ++t)
      aq[m][t] = ldA(qh, DK, q0 + m * 16, t * 32, lane);

  for (int j = 0; j < 8; ++j) {               // wave covers cols [wid*128, +128)
    int cb = wid * 128 + j * 16;
    v8f am[2];
    am[0] = vzero8(); am[1] = vzero8();
#pragma unroll
    for (int t = 0; t < 2; ++t) {             // Dk = 64 => two K=32 steps
      v16bf bk = ldB(kh, DK, cb, t * 32, lane);
      am[0] = wmma_bf16(aq[0][t], bk, am[0]);
      am[1] = wmma_bf16(aq[1][t], bk, am[1]);
    }
    int c = cb + (lane & 15);                 // score column (key index)
#pragma unroll
    for (int m = 0; m < 2; ++m) {
#pragma unroll
      for (int e = 0; e < 8; ++e) {
        int rl = m * 16 + ((lane >> 4) << 3) + e;   // local q row
        int rq = q0 + rl;                           // global q row
        float v = am[m][e] * 0.125f;                // 1/sqrt(64)
        int mk = mb[(size_t)rq * SS + c];
        if (mk == 0) v = -1e9f;
        sp[(size_t)rl * 1024 + c] = __float2bfloat16(v);
      }
    }
  }
  __syncthreads();

  // ---- stage 2: softmax (rows 4*wid .. 4*wid+3; 8 lanes per row) ----------
  {
    int rloc = wid * 4 + (lane >> 3);
    int seg  = lane & 7;
    __hip_bfloat16* rowp = sp + (size_t)rloc * 1024 + seg * 128;

    float mx = -3.0e38f;
    for (int i = 0; i < 128; ++i)
      mx = fmaxf(mx, __bfloat162float(rowp[i]));
    mx = fmaxf(mx, __shfl_xor(mx, 1, 32));
    mx = fmaxf(mx, __shfl_xor(mx, 2, 32));
    mx = fmaxf(mx, __shfl_xor(mx, 4, 32));

    float sum = 0.f;
    for (int i = 0; i < 128; ++i) {
      float e = __expf(__bfloat162float(rowp[i]) - mx);
      sum += e;
      rowp[i] = __float2bfloat16(e);
    }
    sum += __shfl_xor(sum, 1, 32);
    sum += __shfl_xor(sum, 2, 32);
    sum += __shfl_xor(sum, 4, 32);
    float inv = 1.0f / sum;
    for (int i = 0; i < 128; ++i)
      rowp[i] = __float2bfloat16(__bfloat162float(rowp[i]) * inv);
  }
  __syncthreads();

  // ---- stage 3: ctx = P @ v  (one 16x16 output tile per wave) -------------
  {
    int mt = wid >> 2, nt = wid & 3;          // 2 m-tiles x 4 n-tiles
    v8f acc = vzero8();
    for (int k0 = 0; k0 < SS; k0 += 32) {
      v16bf ap = ldA_lds(sp, 1024, mt * 16, k0, lane);
      v16bf bv = ldB(vh, SS, nt * 16, k0, lane);
      acc = wmma_bf16(ap, bv, acc);
    }
    int d = nt * 16 + (lane & 15);
#pragma unroll
    for (int e = 0; e < 8; ++e) {
      int r = q0 + mt * 16 + ((lane >> 4) << 3) + e;
      ctx[((size_t)b * SS + r) * DM + h * DK + d] = __float2bfloat16(acc[e]);
    }
  }
}

// ---------------------------------------------------------------------------
extern "C" void kernel_launch(void* const* d_in, const int* in_sizes, int n_in,
                              void* d_out, int out_size, void* d_ws, size_t ws_size,
                              hipStream_t stream) {
  const float* Q  = (const float*)d_in[0];
  const float* K  = (const float*)d_in[1];
  const float* V  = (const float*)d_in[2];
  const int*   Mk = (const int*)  d_in[3];
  const float* Wq = (const float*)d_in[4];
  const float* bq = (const float*)d_in[5];
  const float* Wk = (const float*)d_in[6];
  const float* bk = (const float*)d_in[7];
  const float* Wv = (const float*)d_in[8];
  const float* bv = (const float*)d_in[9];
  const float* Wo = (const float*)d_in[10];
  const float* bo = (const float*)d_in[11];
  float* out = (float*)d_out;

  const size_t MB = 1ull << 20;
  char* w = (char*)d_ws;
  __hip_bfloat16* Xq  = (__hip_bfloat16*)(w + 0 * MB);    // 8 MB each
  __hip_bfloat16* Xk  = (__hip_bfloat16*)(w + 8 * MB);
  __hip_bfloat16* Xv  = (__hip_bfloat16*)(w + 16 * MB);
  __hip_bfloat16* Wqb = (__hip_bfloat16*)(w + 24 * MB);   // 2 MB each
  __hip_bfloat16* Wkb = (__hip_bfloat16*)(w + 26 * MB);
  __hip_bfloat16* Wvb = (__hip_bfloat16*)(w + 28 * MB);
  __hip_bfloat16* Wob = (__hip_bfloat16*)(w + 30 * MB);
  __hip_bfloat16* qbf = (__hip_bfloat16*)(w + 32 * MB);   // [B,H,S,Dk]
  __hip_bfloat16* kbf = (__hip_bfloat16*)(w + 40 * MB);   // [B,H,S,Dk]
  __hip_bfloat16* vbf = (__hip_bfloat16*)(w + 48 * MB);   // [B,H,Dk,S]
  __hip_bfloat16* cbf = (__hip_bfloat16*)(w + 56 * MB);   // [B*S, DM]

  const int NX = BB * SS * DM;   // 4 Mi elements
  const int NW = DM * DM;        // 1 Mi elements
  f2bf_kernel<<<1024, 256, 0, stream>>>(Q,  Xq,  NX);
  f2bf_kernel<<<1024, 256, 0, stream>>>(K,  Xk,  NX);
  f2bf_kernel<<<1024, 256, 0, stream>>>(V,  Xv,  NX);
  f2bf_kernel<<<512,  256, 0, stream>>>(Wq, Wqb, NW);
  f2bf_kernel<<<512,  256, 0, stream>>>(Wk, Wkb, NW);
  f2bf_kernel<<<512,  256, 0, stream>>>(Wv, Wvb, NW);
  f2bf_kernel<<<512,  256, 0, stream>>>(Wo, Wob, NW);

  const int M = BB * SS;   // 4096
  dim3 ggrid(DM / 128, M / 128);   // (8, 32)
  gemm_nt_bf16<<<ggrid, 256, 0, stream>>>(Xq, Wqb, bq, qbf, M, DM, DM, 0);
  gemm_nt_bf16<<<ggrid, 256, 0, stream>>>(Xk, Wkb, bk, kbf, M, DM, DM, 0);
  gemm_nt_bf16<<<ggrid, 256, 0, stream>>>(Xv, Wvb, bv, vbf, M, DM, DM, 1);

  dim3 agrid(SS / 32, HH, BB);     // (32, 16, 4)
  attn_kernel<<<agrid, 256, 0, stream>>>(qbf, kbf, vbf, Mk, cbf);

  gemm_nt_bf16<<<ggrid, 256, 0, stream>>>(cbf, Wob, bo, out, M, DM, DM, 2);
}
